// BiDirectionalRecurrentNetworkScratch_44908178047581
// MI455X (gfx1250) — matrix operand order repaired and checked by reading, not compiled, not run on hardware
//
#include <hip/hip_runtime.h>
#include <stdint.h>

// ---------------------------------------------------------------------------
// BiDirectional RNN for MI455X (gfx1250): bf16 WMMA + TDM/async-LDS staging.
//   T=4096, IN=1024, H=2048, OUT=1024
// ---------------------------------------------------------------------------

typedef __attribute__((ext_vector_type(16))) __bf16 v16bf;
typedef __attribute__((ext_vector_type(8)))  float  v8f;
typedef unsigned int v4ui __attribute__((ext_vector_type(4)));
typedef int          v8i_ __attribute__((ext_vector_type(8)));
typedef int          v4i_ __attribute__((ext_vector_type(4)));

// Address-space-qualified vector pointers for the async-LDS builtins.
typedef __attribute__((address_space(1))) v4i_ gv4i_t;   // global (AS1)
typedef __attribute__((address_space(3))) v4i_ lv4i_t;   // LDS    (AS3)

#define T_DIM   4096
#define IN_DIM  1024
#define H_DIM   2048
#define OUT_DIM 1024

// --- CDNA5 feature detection (device pass only; host pass parses fallback) --
#if __has_builtin(__builtin_amdgcn_global_load_async_to_lds_b128) && \
    __has_builtin(__builtin_amdgcn_s_wait_asynccnt)
#define HAVE_ASYNC 1
#else
#define HAVE_ASYNC 0
#endif

#if __has_builtin(__builtin_amdgcn_tensor_load_to_lds) && \
    __has_builtin(__builtin_amdgcn_s_wait_tensorcnt)
#define HAVE_TDM 1
#else
#define HAVE_TDM 0
#endif

__device__ __forceinline__ unsigned short f32_to_bf16(float f) {
  unsigned u = __float_as_uint(f);
  unsigned r = u + 0x7fffu + ((u >> 16) & 1u);   // round-to-nearest-even
  return (unsigned short)(r >> 16);
}
__device__ __forceinline__ float bf16_to_f32(unsigned short h) {
  return __uint_as_float(((unsigned)h) << 16);
}

// 16B global -> LDS copy, async (ASYNCcnt) when available.
__device__ __forceinline__ void async_copy16(void* ldsdst, const void* gsrc) {
#if HAVE_ASYNC
  __builtin_amdgcn_global_load_async_to_lds_b128(
      (gv4i_t*)(uintptr_t)gsrc,
      (lv4i_t*)(unsigned)(uintptr_t)ldsdst, 0, 0);
#else
  *(uint4*)ldsdst = *(const uint4*)gsrc;
#endif
}
template <int N> __device__ __forceinline__ void async_wait() {
#if HAVE_ASYNC
  __builtin_amdgcn_s_wait_asynccnt(N);
#endif
}
template <int N> __device__ __forceinline__ void tensor_wait() {
#if HAVE_TDM
  __builtin_amdgcn_s_wait_tensorcnt(N);
#endif
}

#if HAVE_TDM
// TDM 2D tile load: tile_x (elems, bf16) x tile_y rows, row stride = K elems.
// LDS padding via D#: pad_interval enc 3 (16 DW = 64B rows), pad_amount enc 3
// (4 DW = 16B) -> LDS row stride 80B, matching the fragment addressing below.
__device__ __forceinline__ void tdm_load_tile(const void* gsrc, void* ldsdst,
                                              int K_elems, int rows_total,
                                              int tile_x, int tile_y) {
  unsigned long long ga = (unsigned long long)(uintptr_t)gsrc;
  unsigned lds = (unsigned)(uintptr_t)ldsdst;
  v4ui g0;
  g0[0] = 1u;                                   // count=1 (valid, user mode)
  g0[1] = lds;                                  // lds_addr (bytes)
  g0[2] = (unsigned)ga;                         // global_addr[31:0]
  g0[3] = (unsigned)((ga >> 32) & 0x01FFFFFFu)  // global_addr[56:32]
          | (2u << 30);                         // type=2 ("image")
  v8i_ g1;
  g1[0] = (int)((1u << 16)        // data_size=1 -> 2 bytes
                | (1u << 20)      // pad_enable
                | (3u << 22)      // pad_interval: 16 DWORDs
                | (3u << 25));    // pad_amount: 4 DWORDs
  g1[1] = (int)(((unsigned)K_elems & 0xFFFFu) << 16);         // tensor_dim0 lo
  g1[2] = (int)((((unsigned)K_elems >> 16) & 0xFFFFu) |
                (((unsigned)rows_total & 0xFFFFu) << 16));    // d0 hi | d1 lo
  g1[3] = (int)(((((unsigned)rows_total) >> 16) & 0xFFFFu) |
                (((unsigned)tile_x & 0xFFFFu) << 16));        // d1 hi | tile0
  g1[4] = (int)((unsigned)tile_y & 0xFFFFu);                  // tile1 (tile2=0)
  g1[5] = (int)(unsigned)K_elems;                             // d0_stride lo32
  g1[6] = 0;                                                  // d0 hi | d1 lo
  g1[7] = 0;
  v4i_ zz4 = {0, 0, 0, 0};                      // groups 2/3 unused (2D tile)
  v8i_ zz8 = {0, 0, 0, 0, 0, 0, 0, 0};
  __builtin_amdgcn_tensor_load_to_lds(g0, g1, zz4, zz4, zz8, 0);
}
#endif

// ---------------------------------------------------------------------------
// f32 -> bf16 conversion (grid-stride)
// ---------------------------------------------------------------------------
__global__ void cvt_f32_bf16(const float* __restrict__ src,
                             unsigned short* __restrict__ dst, int n) {
  int i = blockIdx.x * blockDim.x + threadIdx.x;
  int stride = gridDim.x * blockDim.x;
  for (; i < n; i += stride) dst[i] = f32_to_bf16(src[i]);
}

__global__ void init_counters(unsigned* __restrict__ c) {
  if (threadIdx.x < 2) c[threadIdx.x] = 0u;
}

// ---------------------------------------------------------------------------
// WMMA GEMM: C[M,N] = A[M,K](bf16,row-major) x W[N,K](bf16,row-major)^T
//            (+ bias[n]) (+ accumulate into outF)   out: f32 or bf16
// block = 256 threads (8 waves), tile 128x128, K-step 32, double-buffered
// LDS staging via TDM (wave 0 issues 2 tensor loads/tile, TENSORcnt-paced).
// ---------------------------------------------------------------------------
#define BM 128
#define BN 128
#define BK 32
#define LDST 40   // padded LDS row stride (elements) = 80B, conflict-free

__launch_bounds__(256)
__global__ void wmma_gemm_bf16(const unsigned short* __restrict__ A,
                               const unsigned short* __restrict__ W,
                               const float* __restrict__ bias,
                               float* __restrict__ outF,
                               unsigned short* __restrict__ outH,
                               int M, int N, int K, int accumulate) {
  __shared__ __align__(16) unsigned short sA[2][BM * LDST];
  __shared__ __align__(16) unsigned short sB[2][BN * LDST];

  const int tid   = threadIdx.x;
  const int lane  = tid & 31;
  const int wave  = tid >> 5;      // 0..7
  const int waveM = wave >> 2;     // 0..1 -> 64 rows each
  const int waveN = wave & 3;      // 0..3 -> 32 cols each
  const int m0 = blockIdx.y * BM;
  const int n0 = blockIdx.x * BN;
  const int half = lane >> 4;      // 0/1
  const int l16  = lane & 15;

  v8f acc[4][2];
  const v8f vzero = {0.f, 0.f, 0.f, 0.f, 0.f, 0.f, 0.f, 0.f};
#pragma unroll
  for (int i = 0; i < 4; ++i)
#pragma unroll
    for (int j = 0; j < 2; ++j) acc[i][j] = vzero;

  union UB { v16bf v; unsigned u[8]; };

  // Stage one 128x32 A-tile + 128x32 B-tile into buffer `buf`.
  auto stage = [&](int buf, int k0) {
#if HAVE_TDM
    if (wave == 0) {
      tdm_load_tile(A + (size_t)m0 * K + k0, &sA[buf][0], K, M, BK, BM);
      tdm_load_tile(W + (size_t)n0 * K + k0, &sB[buf][0], K, N, BK, BN);
    }
#else
#pragma unroll
    for (int i = 0; i < 2; ++i) {
      int c    = tid + i * 256;
      int row  = c >> 2;
      int col8 = (c & 3) * 8;
      async_copy16(&sA[buf][row * LDST + col8],
                   A + (size_t)(m0 + row) * K + k0 + col8);
      async_copy16(&sB[buf][row * LDST + col8],
                   W + (size_t)(n0 + row) * K + k0 + col8);
    }
#endif
  };

  const int nk = K / BK;
  stage(0, 0);

  for (int kt = 0; kt < nk; ++kt) {
    const int cur = kt & 1;
    if (kt + 1 < nk) stage(cur ^ 1, (kt + 1) * BK);  // DMA next tile
#if HAVE_TDM
    if (kt + 1 < nk) tensor_wait<2>(); else tensor_wait<0>();
#else
    if (kt + 1 < nk) async_wait<4>(); else async_wait<0>();
#endif
    __syncthreads();

    const unsigned short* pA = &sA[cur][0];
    const unsigned short* pB = &sB[cur][0];

    // B fragments (32x16, K x N): lane = col n, VGPR v -> K = half*16 + 2v
    UB bfr[2];
#pragma unroll
    for (int nt = 0; nt < 2; ++nt) {
      int n = waveN * 32 + nt * 16 + l16;
#pragma unroll
      for (int v = 0; v < 8; ++v) {
        int kk = half * 16 + v * 2;
        bfr[nt].u[v] = *(const unsigned*)(pB + n * LDST + kk);
      }
    }
    // A fragments (16x32, M x K): lane%16 = row m,
    // VGPR v -> K = (lane/16)*16 + 2v
#pragma unroll
    for (int mt = 0; mt < 4; ++mt) {
      UB afr;
      int m = waveM * 64 + mt * 16 + l16;
#pragma unroll
      for (int v = 0; v < 8; ++v) {
        int kk = half * 16 + v * 2;
        afr.u[v] = *(const unsigned*)(pA + m * LDST + kk);
      }
#pragma unroll
      for (int nt = 0; nt < 2; ++nt) {
        acc[mt][nt] = __builtin_amdgcn_wmma_f32_16x16x32_bf16(
            false, afr.v, false, bfr[nt].v, (short)0, acc[mt][nt],
            false, false);
      }
    }
    __syncthreads();   // everyone done with `cur` before it is re-staged
  }

  // Epilogue: D element r of lane L -> m = r + 8*(L/16), n = L%16
#pragma unroll
  for (int mt = 0; mt < 4; ++mt)
#pragma unroll
    for (int nt = 0; nt < 2; ++nt) {
      int n = n0 + waveN * 32 + nt * 16 + l16;
      float bv = bias ? bias[n] : 0.0f;
#pragma unroll
      for (int r = 0; r < 8; ++r) {
        int m = m0 + waveM * 64 + mt * 16 + half * 8 + r;
        size_t idx = (size_t)m * N + n;
        float val = acc[mt][nt][r] + bv;
        if (outF) {
          if (accumulate) val += outF[idx];
          outF[idx] = val;
        } else {
          outH[idx] = f32_to_bf16(val);
        }
      }
    }
}

// ---------------------------------------------------------------------------
// Recurrent scan: h_t = tanh(xw_t + Wh h_{t-1}), both directions.
// Grid = 256 WGs: blocks [0,128) forward, [128,256) backward.
// Each WG owns 16 rows of h; Wh slice (16x2048 bf16, padded) cached in LDS
// via async-to-LDS copies.  Wave 0 does the GEMV via WMMA: h broadcast into
// all B columns => every D column equals Wh_slice * h.  Grid-wide per-step
// sync via global counter (release fence + atomicAdd, acquire spin).
// ---------------------------------------------------------------------------
#define SCAN_WGS 128
#define ROWS_PER_WG 16              // H_DIM / SCAN_WGS
#define WH_LDST 2056                // padded stride: conflict-free, 16B-aligned

__launch_bounds__(256)
__global__ void scan_kernel(const unsigned short* __restrict__ Whf,
                            const unsigned short* __restrict__ Whb,
                            const unsigned short* __restrict__ xwf,
                            const unsigned short* __restrict__ xwb,
                            unsigned short* __restrict__ hf,
                            unsigned short* __restrict__ hb,
                            unsigned int* __restrict__ counters) {
  __shared__ __align__(16) unsigned short whs[ROWS_PER_WG * WH_LDST];
  __shared__ __align__(16) unsigned short hs[H_DIM];

  const int dir = (blockIdx.x >= SCAN_WGS) ? 1 : 0;
  const int wg  = blockIdx.x & (SCAN_WGS - 1);
  const unsigned short* Wh   = dir ? Whb : Whf;
  const unsigned short* xw   = dir ? xwb : xwf;
  unsigned short*       hist = dir ? hb  : hf;
  unsigned int*         ctr  = counters + dir;

  const int tid  = threadIdx.x;
  const int lane = tid & 31;
  const int half = lane >> 4;
  const int l16  = lane & 15;
  const int r0   = wg * ROWS_PER_WG;

  // Preload this WG's Wh slice into LDS (16 x 2048 bf16, async copies).
  for (int i = tid; i < ROWS_PER_WG * (H_DIM / 8); i += 256) {
    int row  = i / (H_DIM / 8);
    int col8 = (i % (H_DIM / 8)) * 8;
    async_copy16(whs + row * WH_LDST + col8,
                 Wh + (size_t)(r0 + row) * H_DIM + col8);
  }
  async_wait<0>();
  __syncthreads();

  union UB { v16bf v; unsigned u[8]; };

#pragma unroll 1
  for (int s = 0; s < T_DIM; ++s) {
    const int t     = dir ? (T_DIM - 1 - s) : s;
    const int tprev = dir ? (t + 1) : (t - 1);

    // Stage h_{t-1} (bf16, 4KB) into LDS; zeros on first step.
    if (s == 0) {
      *(uint4*)(hs + tid * 8) = make_uint4(0u, 0u, 0u, 0u);
    } else {
      async_copy16(hs + tid * 8, hist + (size_t)tprev * H_DIM + tid * 8);
      async_wait<0>();
    }
    __syncthreads();

    if (tid < 32) {
      v8f acc = {0.f, 0.f, 0.f, 0.f, 0.f, 0.f, 0.f, 0.f};
#pragma unroll 4
      for (int k0 = 0; k0 < H_DIM; k0 += 32) {
        UB a, b;
#pragma unroll
        for (int v = 0; v < 8; ++v) {
          int ka = half * 16 + v * 2;
          a.u[v] = *(const unsigned*)(whs + l16 * WH_LDST + k0 + ka);
          int kb = half * 16 + v * 2;           // same h for every column n
          b.u[v] = *(const unsigned*)(hs + k0 + kb);
        }
        acc = __builtin_amdgcn_wmma_f32_16x16x32_bf16(
            false, a.v, false, b.v, (short)0, acc, false, false);
      }
      // Every D column equals the GEMV; lanes with n==0 (lanes 0,16) write.
      if (l16 == 0) {
#pragma unroll
        for (int r = 0; r < 8; ++r) {
          int m = r0 + half * 8 + r;
          float pre = bf16_to_f32(xw[(size_t)t * H_DIM + m]) + acc[r];
          hist[(size_t)t * H_DIM + m] = f32_to_bf16(tanhf(pre));
        }
      }
    }

    // Release our h_t, then wait for all WGs of this direction.
    if (tid == 0) {
      __threadfence();
      atomicAdd(ctr, 1u);
      unsigned target = (unsigned)(s + 1) * SCAN_WGS;
      while (__hip_atomic_load(ctr, __ATOMIC_ACQUIRE,
                               __HIP_MEMORY_SCOPE_AGENT) < target) {
        __builtin_amdgcn_s_sleep(2);
      }
    }
    __syncthreads();
  }
}

// ---------------------------------------------------------------------------
// Host-side orchestration
// ---------------------------------------------------------------------------
extern "C" void kernel_launch(void* const* d_in, const int* in_sizes, int n_in,
                              void* d_out, int out_size, void* d_ws,
                              size_t ws_size, hipStream_t stream) {
  (void)in_sizes; (void)n_in; (void)out_size; (void)ws_size;

  const float* x    = (const float*)d_in[0];
  const float* Wx_f = (const float*)d_in[1];
  const float* Wh_f = (const float*)d_in[2];
  const float* bh_f = (const float*)d_in[3];
  const float* Wx_b = (const float*)d_in[4];
  const float* Wh_b = (const float*)d_in[5];
  const float* bh_b = (const float*)d_in[6];
  const float* Wy_f = (const float*)d_in[7];
  const float* Wy_b = (const float*)d_in[8];
  const float* by   = (const float*)d_in[9];
  float* y = (float*)d_out;

  char* ws = (char*)d_ws;
  size_t off = 0;
  auto alloc = [&](size_t bytes) -> char* {
    char* p = ws + off;
    off += (bytes + 255) & ~(size_t)255;
    return p;
  };
  unsigned short* xw_f   = (unsigned short*)alloc((size_t)T_DIM * H_DIM * 2);
  unsigned short* xw_b   = (unsigned short*)alloc((size_t)T_DIM * H_DIM * 2);
  unsigned short* hf_bf  = (unsigned short*)alloc((size_t)T_DIM * H_DIM * 2);
  unsigned short* hb_bf  = (unsigned short*)alloc((size_t)T_DIM * H_DIM * 2);
  unsigned short* x_bf   = (unsigned short*)alloc((size_t)T_DIM * IN_DIM * 2);
  unsigned short* Wxf_bf = (unsigned short*)alloc((size_t)H_DIM * IN_DIM * 2);
  unsigned short* Wxb_bf = (unsigned short*)alloc((size_t)H_DIM * IN_DIM * 2);
  unsigned short* Whf_bf = (unsigned short*)alloc((size_t)H_DIM * H_DIM * 2);
  unsigned short* Whb_bf = (unsigned short*)alloc((size_t)H_DIM * H_DIM * 2);
  unsigned short* Wyf_bf = (unsigned short*)alloc((size_t)OUT_DIM * H_DIM * 2);
  unsigned short* Wyb_bf = (unsigned short*)alloc((size_t)OUT_DIM * H_DIM * 2);
  unsigned int*   ctrs   = (unsigned int*)alloc(256);

  init_counters<<<1, 32, 0, stream>>>(ctrs);

  // f32 -> bf16 conversions
  cvt_f32_bf16<<<1024, 256, 0, stream>>>(x,    x_bf,   T_DIM * IN_DIM);
  cvt_f32_bf16<<<512,  256, 0, stream>>>(Wx_f, Wxf_bf, H_DIM * IN_DIM);
  cvt_f32_bf16<<<512,  256, 0, stream>>>(Wx_b, Wxb_bf, H_DIM * IN_DIM);
  cvt_f32_bf16<<<1024, 256, 0, stream>>>(Wh_f, Whf_bf, H_DIM * H_DIM);
  cvt_f32_bf16<<<1024, 256, 0, stream>>>(Wh_b, Whb_bf, H_DIM * H_DIM);
  cvt_f32_bf16<<<512,  256, 0, stream>>>(Wy_f, Wyf_bf, OUT_DIM * H_DIM);
  cvt_f32_bf16<<<512,  256, 0, stream>>>(Wy_b, Wyb_bf, OUT_DIM * H_DIM);

  // Input projections: xw = x * Wx^T + bh  -> bf16 [T, H]
  dim3 g1(H_DIM / BN, T_DIM / BM);
  wmma_gemm_bf16<<<g1, 256, 0, stream>>>(x_bf, Wxf_bf, bh_f, nullptr, xw_f,
                                         T_DIM, H_DIM, IN_DIM, 0);
  wmma_gemm_bf16<<<g1, 256, 0, stream>>>(x_bf, Wxb_bf, bh_b, nullptr, xw_b,
                                         T_DIM, H_DIM, IN_DIM, 0);

  // Bidirectional recurrent scan (grid-synced, WMMA GEMV per step)
  scan_kernel<<<2 * SCAN_WGS, 256, 0, stream>>>(Whf_bf, Whb_bf, xw_f, xw_b,
                                                hf_bf, hb_bf, ctrs);

  // Output projection: y = hf * Wy_f^T + by, then y += hb * Wy_b^T
  dim3 g2(OUT_DIM / BN, T_DIM / BM);
  wmma_gemm_bf16<<<g2, 256, 0, stream>>>(hf_bf, Wyf_bf, by, y, nullptr,
                                         T_DIM, OUT_DIM, H_DIM, 0);
  wmma_gemm_bf16<<<g2, 256, 0, stream>>>(hb_bf, Wyb_bf, nullptr, y, nullptr,
                                         T_DIM, OUT_DIM, H_DIM, 1);
}
